// DCDL_8031588843765
// MI455X (gfx1250) — matrix operand order repaired and checked by reading, not compiled, not run on hardware
//
#include <hip/hip_runtime.h>
#include <hip/hip_bf16.h>
#include <math.h>

// ---------------- problem constants (match reference) ----------------
constexpr int U_ = 20000, B_ = 15000, I_ = 40000, EMB = 64, HID = 256;
constexpr int N_ = U_ + B_;
constexpr int TIMESTEPS = 15;
constexpr int BATCH = 4096, NB = 100;
constexpr float L2NORM = 1e-5f;

// ---------------- float workspace layout (float offsets) ----------------
constexpr long O_WX     = 0;                              // N x 64
constexpr long O_GAT    = O_WX     + (long)N_ * EMB;      // N x 64
constexpr long O_EMBR   = O_GAT    + (long)N_ * EMB;      // N x 64
constexpr long O_IUI    = O_EMBR   + (long)N_ * EMB;      // I x 64
constexpr long O_IBI    = O_IUI    + (long)I_ * EMB;      // I x 64
constexpr long O_ITEMSF = O_IBI    + (long)I_ * EMB;      // I x 64
constexpr long O_UBUF   = O_ITEMSF + (long)I_ * EMB;      // U x 64
constexpr long O_BBUF   = O_UBUF   + (long)U_ * EMB;      // B x 64
constexpr long O_USERSF = O_BBUF   + (long)B_ * EMB;      // U x 64
constexpr long O_BUNDF  = O_USERSF + (long)U_ * EMB;      // B x 64
constexpr long O_PREDX  = O_BUNDF  + (long)B_ * EMB;      // U x 64
constexpr long O_EL     = O_PREDX  + (long)U_ * EMB;      // N
constexpr long O_ER     = O_EL + N_;                      // N
constexpr long O_M      = O_ER + N_;                      // N
constexpr long O_S      = O_M  + N_;                      // N
constexpr long O_EX     = O_S  + N_;                      // Eg (= N*17)
constexpr long O_FLOAT_END = ((O_EX + (long)N_ * 17 + 7) / 8) * 8;  // 32B aligned

// ---------------- bf16 workspace layout (bf16-element offsets) ----------
// sizes: packed A buffers are roundup(M,16)*K elements
constexpr long PB_E0P  = 0;                                        // 2188*16*64
constexpr long PB_HINP = PB_E0P  + (long)2188 * 16 * 64;           // 1250*16*192
constexpr long PB_H1P  = PB_HINP + (long)1250 * 16 * 192;          // 1250*16*256
constexpr long PB_WGP  = PB_H1P  + (long)1250 * 16 * 256;          // 64*64
constexpr long PB_W1P  = PB_WGP  + 64 * 64;                        // 192*256
constexpr long PB_W2P  = PB_W1P  + 192 * 256;                      // 256*64

// ---------------- output layout (float offsets) ----------------
constexpr long OUT_PRED  = 0;
constexpr long OUT_BOUND = (long)BATCH * NB;   // 409600
constexpr long OUT_REG   = OUT_BOUND + BATCH;  // 413696
constexpr long OUT_DIFF  = OUT_REG + 1;
constexpr long OUT_UF    = OUT_DIFF + 1;

// ---------------- WMMA types ----------------
typedef __attribute__((ext_vector_type(16))) __bf16 bf16x16;
typedef __attribute__((ext_vector_type(8)))  float  f32x8;

// Fragment-layout index helpers (from CDNA5 ISA 16-bit WMMA VGPR tables).
// A fragment (16x32): lane l holds row M=l&15; element p <-> K = (p>>3)*16 + (l>>4)*8 + (p&7)
__device__ inline long a_pack_idx(int row, int k, int K) {
    int mt = row >> 4, l16 = row & 15;
    int ks = k >> 5,  kk  = k & 31;
    int hi = kk >> 4;           // K>=16 within the 32-step -> p upper half
    int kk2 = kk & 15;
    int half = kk2 >> 3;        // lane half
    int p = (kk2 & 7) + (hi << 3);
    int lane = (half << 4) + l16;
    return (((long)mt * (K >> 5) + ks) * 32 + lane) * 16 + p;
}
// B fragment (32x16): lane l holds col N=l&15; element p <-> K = (p>>3)*16 + (l>>4)*8 + (p&7)
__device__ inline long b_pack_idx(int k, int n, int K) {
    int nt = n >> 4, l16 = n & 15;
    int ks = k >> 5, kk = k & 31;
    int half = (kk & 15) >> 3;
    int p = (kk & 7) + ((kk >> 4) << 3);
    int lane = (half << 4) + l16;
    return (((long)nt * (K >> 5) + ks) * 32 + lane) * 16 + p;
}

// =====================================================================
// GEMM on pre-packed bf16 fragments.
// blockDim = 128 (4 waves); wave computes a 16x64 output strip.
// Per k-step: 1 A-fragment vector load + 4 B loads + 4 independent WMMAs.
// Output: f32 row-major (Cf) or packed-A bf16 for the next GEMM (Cp).
// =====================================================================
__global__ __launch_bounds__(128) void gemm_wmma_packed(
    const __bf16* __restrict__ Ap, const __bf16* __restrict__ Wp,
    const float* __restrict__ bias, float* __restrict__ Cf,
    __bf16* __restrict__ Cp, int M, int K, int Nc, int act)
{
    const int lane = threadIdx.x & 31;
    const int wave = threadIdx.x >> 5;
    const int half = lane >> 4;
    const int l16  = lane & 15;
    const int tileM = blockIdx.x * 64 + wave * 16;
    if (tileM >= M) return;            // wave-uniform guard (EXEC all-ones at WMMA)
    const int tileN = blockIdx.y * 64;
    const int nt0   = tileN >> 4;
    const int ksteps = K >> 5;

    const bf16x16* __restrict__ Av = (const bf16x16*)Ap;
    const bf16x16* __restrict__ Wv = (const bf16x16*)Wp;
    const long aBase = (long)(tileM >> 4) * ksteps * 32 + lane;

    f32x8 acc[4] = {};
#pragma unroll 2
    for (int ks = 0; ks < ksteps; ++ks) {
        bf16x16 a = Av[aBase + (long)ks * 32];
#pragma unroll
        for (int tn = 0; tn < 4; ++tn) {
            bf16x16 b = Wv[(((long)(nt0 + tn)) * ksteps + ks) * 32 + lane];
            acc[tn] = __builtin_amdgcn_wmma_f32_16x16x32_bf16(
                          false, a, false, b, (short)0, acc[tn], false, false);
        }
    }

    const int col0 = tileN + l16;
#pragma unroll
    for (int tn = 0; tn < 4; ++tn) {
        const int col = col0 + tn * 16;
        const float bv = bias ? bias[col] : 0.0f;
#pragma unroll
        for (int i = 0; i < 8; ++i) {
            const int r = tileM + i + (half << 3);  // D: VGPR i -> M = i + 8*half
            if (r < M) {
                float v = acc[tn][i] + bv;
                if (act == 1) v = tanhf(v);
                if (Cf) Cf[(long)r * Nc + col] = v;
                if (Cp) Cp[a_pack_idx(r, col, Nc)] = (__bf16)v;
            }
        }
    }
}

// =====================================================================
// Packing kernels
// =====================================================================
// Pack weight W[K x Nc] (f32 row-major) into B-fragment bf16 layout.
__global__ void pack_w_kernel(const float* __restrict__ W, __bf16* __restrict__ Wp,
                              int K, int Nc) {
    long i = (long)blockIdx.x * blockDim.x + threadIdx.x;
    if (i >= (long)K * Nc) return;
    int k = (int)(i / Nc), n = (int)(i % Nc);
    Wp[b_pack_idx(k, n, K)] = (__bf16)W[i];
}

// embed0 = [users_feature; bundles_feature], written straight into packed-A bf16.
__global__ void concat_pack_kernel(const float* __restrict__ uf, const float* __restrict__ bf,
                                   __bf16* __restrict__ e0p, long nu, long ntot) {
    long i = (long)blockIdx.x * blockDim.x + threadIdx.x;
    if (i >= ntot) return;
    int row = (int)(i >> 6), k = (int)(i & 63);
    float v = (i < nu) ? uf[i] : bf[i - nu];
    e0p[a_pack_idx(row, k, EMB)] = (__bf16)v;
}

// Build Hin[U x 192] = [x_noisy | users_f | time_embedding] in packed-A bf16.
__global__ void diffusion_prep_pack(const float* __restrict__ xstart,
                                    const float* __restrict__ noise,
                                    const float* __restrict__ usersf,
                                    const int* __restrict__ t,
                                    __bf16* __restrict__ Hp, long n /* U*192 */) {
    long i = (long)blockIdx.x * blockDim.x + threadIdx.x;
    if (i >= n) return;
    int u = (int)(i / 192);
    int k = (int)(i % 192);
    int tt = t[u];
    float val;
    if (k < 64) {
        float ac = 1.0f;
        for (int s = 0; s <= tt; ++s) {
            float beta = 1e-4f + (0.02f - 1e-4f) * (float)s / (float)(TIMESTEPS - 1);
            ac *= (1.0f - beta);
        }
        long idx = (long)u * EMB + k;
        val = sqrtf(ac) * xstart[idx] + sqrtf(1.0f - ac) * noise[idx];
    } else if (k < 128) {
        val = usersf[(long)u * EMB + (k - 64)];
    } else {
        int f = k - 128;
        float tf = (float)tt;
        if (f < 32) {
            float freq = __expf(-logf(10000.0f) * (float)f / 32.0f);
            val = __sinf(tf * freq);
        } else {
            float freq = __expf(-logf(10000.0f) * (float)(f - 32) / 32.0f);
            val = __cosf(tf * freq);
        }
    }
    Hp[a_pack_idx(u, k, 3 * EMB)] = (__bf16)val;
}

// =====================================================================
// Small utility kernels
// =====================================================================
__global__ void fill_kernel(float* p, float v, long n) {
    long i = (long)blockIdx.x * blockDim.x + threadIdx.x;
    if (i < n) p[i] = v;
}

__global__ void copy_kernel(const float* __restrict__ a, float* __restrict__ o, long n) {
    long i = (long)blockIdx.x * blockDim.x + threadIdx.x;
    if (i < n) o[i] = a[i];
}

// el/er: one wave per row, dot of 64-wide row with attn vectors
__global__ void elr_kernel(const float* __restrict__ Wx, const float* __restrict__ al,
                           const float* __restrict__ ar, float* __restrict__ el,
                           float* __restrict__ er, int n) {
    int w = (int)(((long)blockIdx.x * blockDim.x + threadIdx.x) >> 5);
    int lane = threadIdx.x & 31;
    if (w >= n) return;
    const float* x = Wx + (long)w * EMB;
    float a = x[lane] * al[lane] + x[lane + 32] * al[lane + 32];
    float b = x[lane] * ar[lane] + x[lane + 32] * ar[lane + 32];
    for (int off = 16; off; off >>= 1) {
        a += __shfl_down(a, off, 32);
        b += __shfl_down(b, off, 32);
    }
    if (lane == 0) { el[w] = a; er[w] = b; }
}

__device__ inline void atomicMaxFloat(float* addr, float val) {
    if (val >= 0.0f) atomicMax((int*)addr, __float_as_int(val));
    else             atomicMin((unsigned int*)addr, __float_as_uint(val));
}

// pass A: e = leaky_relu(el[src]+er[dst]); segment max into m[dst]
__global__ void gat_edge_e(const float* __restrict__ el, const float* __restrict__ er,
                           const int* __restrict__ src, const int* __restrict__ dst,
                           float* __restrict__ e, float* __restrict__ m, long E) {
    long i = (long)blockIdx.x * blockDim.x + threadIdx.x;
    if (i >= E) return;
    float v = el[src[i]] + er[dst[i]];
    v = v > 0.0f ? v : 0.2f * v;
    e[i] = v;
    atomicMaxFloat(&m[dst[i]], v);
}

// pass B: ex = exp(e - m[dst]); s[dst] += ex   (ex overwrites e)
__global__ void gat_edge_ex(float* __restrict__ e, const int* __restrict__ dst,
                            const float* __restrict__ m, float* __restrict__ s, long E) {
    long i = (long)blockIdx.x * blockDim.x + threadIdx.x;
    if (i >= E) return;
    float ex = __expf(e[i] - m[dst[i]]);
    e[i] = ex;
    atomicAdd(&s[dst[i]], ex);
}

// pass C: gat[dst,f] += (ex/(s[dst]+1e-9)) * Wx[src,f]   (thread per edge*feature)
__global__ void gat_edge_scatter(const float* __restrict__ ex, const float* __restrict__ s,
                                 const int* __restrict__ src, const int* __restrict__ dst,
                                 const float* __restrict__ Wx, float* __restrict__ gat, long E) {
    long idx = (long)blockIdx.x * blockDim.x + threadIdx.x;
    long edge = idx >> 6;
    int  f    = (int)(idx & 63);
    if (edge >= E) return;
    int d = dst[edge], sr = src[edge];
    float alpha = ex[edge] / (s[d] + 1e-9f);
    atomicAdd(&gat[(long)d * EMB + f], alpha * Wx[(long)sr * EMB + f]);
}

// generic SpMM scatter: out[row,f] += val * (X[col,f] + xbias[f])
__global__ void spmm_scatter(const int* __restrict__ row, const int* __restrict__ col,
                             const float* __restrict__ val, const float* __restrict__ X,
                             const float* __restrict__ xbias, float* __restrict__ out, long E) {
    long idx = (long)blockIdx.x * blockDim.x + threadIdx.x;
    long edge = idx >> 6;
    int  f    = (int)(idx & 63);
    if (edge >= E) return;
    float x = X[(long)col[edge] * EMB + f];
    if (xbias) x += xbias[f];
    atomicAdd(&out[(long)row[edge] * EMB + f], val[edge] * x);
}

// items_f = relu(iui) + relu(ibi) + items_feature
__global__ void items_combine(const float* __restrict__ a, const float* __restrict__ b,
                              const float* __restrict__ base, float* __restrict__ o, long n) {
    long i = (long)blockIdx.x * blockDim.x + threadIdx.x;
    if (i >= n) return;
    float x = a[i], y = b[i];
    o[i] = (x > 0 ? x : 0) + (y > 0 ? y : 0) + base[i];
}

// o = relu(a) + b
__global__ void relu_add(const float* __restrict__ a, const float* __restrict__ b,
                         float* __restrict__ o, long n) {
    long i = (long)blockIdx.x * blockDim.x + threadIdx.x;
    if (i >= n) return;
    float x = a[i];
    o[i] = (x > 0 ? x : 0) + b[i];
}

// sum(x^2) * scale  -> atomicAdd into *out
__global__ void sqsum_kernel(const float* __restrict__ x, long n, float scale, float* out) {
    __shared__ float sm[256];
    long i = (long)blockIdx.x * blockDim.x + threadIdx.x;
    float v = 0.0f;
    if (i < n) { float t = x[i]; v = t * t; }
    sm[threadIdx.x] = v;
    __syncthreads();
    for (int s = 128; s > 0; s >>= 1) {
        if ((int)threadIdx.x < s) sm[threadIdx.x] += sm[threadIdx.x + s];
        __syncthreads();
    }
    if (threadIdx.x == 0) atomicAdd(out, sm[0] * scale);
}

// mean((a-b)^2) contribution -> atomicAdd into *out
__global__ void diffloss_kernel(const float* __restrict__ a, const float* __restrict__ b,
                                long n, float scale, float* out) {
    __shared__ float sm[256];
    long i = (long)blockIdx.x * blockDim.x + threadIdx.x;
    float v = 0.0f;
    if (i < n) { float d = a[i] - b[i]; v = d * d; }
    sm[threadIdx.x] = v;
    __syncthreads();
    for (int s = 128; s > 0; s >>= 1) {
        if ((int)threadIdx.x < s) sm[threadIdx.x] += sm[threadIdx.x + s];
        __syncthreads();
    }
    if (threadIdx.x == 0) atomicAdd(out, sm[0] * scale);
}

// pred[b, j] = dot64(users_f[users[b]], bundles_f[bundles[b, j]]) — wave per output
__global__ void pred_kernel(const float* __restrict__ uf, const float* __restrict__ bf,
                            const int* __restrict__ users, const int* __restrict__ bundles,
                            float* __restrict__ pred) {
    long w = ((long)blockIdx.x * blockDim.x + threadIdx.x) >> 5;
    int lane = threadIdx.x & 31;
    if (w >= (long)BATCH * NB) return;
    int b = (int)(w / NB), j = (int)(w % NB);
    int u  = users[b];
    int bd = bundles[(long)b * NB + j];
    const float* ur = uf + (long)u * EMB;
    const float* br = bf + (long)bd * EMB;
    float s = ur[lane] * br[lane] + ur[lane + 32] * br[lane + 32];
    for (int off = 16; off; off >>= 1) s += __shfl_down(s, off, 32);
    if (lane == 0) pred[w] = s;
}

// user_score_bound[b] = dot64(predicted_x[users[b]], user_bound) — wave per b
__global__ void bound_kernel(const float* __restrict__ px, const float* __restrict__ ub,
                             const int* __restrict__ users, float* __restrict__ out) {
    long w = ((long)blockIdx.x * blockDim.x + threadIdx.x) >> 5;
    int lane = threadIdx.x & 31;
    if (w >= BATCH) return;
    const float* r = px + (long)users[w] * EMB;
    float s = r[lane] * ub[lane] + r[lane + 32] * ub[lane + 32];
    for (int off = 16; off; off >>= 1) s += __shfl_down(s, off, 32);
    if (lane == 0) out[w] = s;
}

// =====================================================================
// Launch
// =====================================================================
static inline int cdivl(long a, long b) { return (int)((a + b - 1) / b); }

extern "C" void kernel_launch(void* const* d_in, const int* in_sizes, int n_in,
                              void* d_out, int out_size, void* d_ws, size_t ws_size,
                              hipStream_t stream) {
    const float* users_feature   = (const float*)d_in[0];
    const float* bundles_feature = (const float*)d_in[1];
    const float* items_feature   = (const float*)d_in[2];
    const float* W_gat   = (const float*)d_in[3];
    const float* attn_l  = (const float*)d_in[4];
    const float* attn_r  = (const float*)d_in[5];
    const float* bias_gat= (const float*)d_in[6];
    const float* W1      = (const float*)d_in[7];
    const float* b1      = (const float*)d_in[8];
    const float* W2      = (const float*)d_in[9];
    const float* b2      = (const float*)d_in[10];
    const float* user_bound = (const float*)d_in[11];
    const float* noise   = (const float*)d_in[12];
    const int*   r_row   = (const int*)d_in[13];
    const int*   r_col   = (const int*)d_in[14];
    const float* r_val   = (const float*)d_in[15];
    const int*   Ai_row  = (const int*)d_in[16];
    const int*   Ai_col  = (const int*)d_in[17];
    const float* Ai_val  = (const float*)d_in[18];
    const int*   Bi_row  = (const int*)d_in[19];
    const int*   Bi_col  = (const int*)d_in[20];
    const float* Bi_val  = (const float*)d_in[21];
    const int*   ui_row  = (const int*)d_in[22];
    const int*   ui_col  = (const int*)d_in[23];
    const float* ui_val  = (const float*)d_in[24];
    const int*   bi_row  = (const int*)d_in[25];
    const int*   bi_col  = (const int*)d_in[26];
    const float* bi_val  = (const float*)d_in[27];
    const int*   g_src   = (const int*)d_in[28];
    const int*   g_dst   = (const int*)d_in[29];
    const int*   users   = (const int*)d_in[30];
    const int*   bundles = (const int*)d_in[31];
    const int*   t       = (const int*)d_in[32];

    const long Er  = in_sizes[13];
    const long Eai = in_sizes[16];
    const long Ebi = in_sizes[19];
    const long Eui = in_sizes[22];
    const long Ebb = in_sizes[25];
    const long Eg  = in_sizes[28];

    float*  ws  = (float*)d_ws;
    __bf16* wsb = (__bf16*)(ws + O_FLOAT_END);
    float*  out = (float*)d_out;

    const int T = 256;

    // ---- zero / init accumulators ----
    fill_kernel<<<cdivl((long)N_,       T), T, 0, stream>>>(ws + O_M,   -INFINITY, N_);
    fill_kernel<<<cdivl((long)N_,       T), T, 0, stream>>>(ws + O_S,    0.0f, N_);
    fill_kernel<<<cdivl((long)N_*EMB,   T), T, 0, stream>>>(ws + O_GAT,  0.0f, (long)N_*EMB);
    fill_kernel<<<cdivl((long)N_*EMB,   T), T, 0, stream>>>(ws + O_EMBR, 0.0f, (long)N_*EMB);
    fill_kernel<<<cdivl((long)I_*EMB,   T), T, 0, stream>>>(ws + O_IUI,  0.0f, (long)I_*EMB);
    fill_kernel<<<cdivl((long)I_*EMB,   T), T, 0, stream>>>(ws + O_IBI,  0.0f, (long)I_*EMB);
    fill_kernel<<<cdivl((long)U_*EMB,   T), T, 0, stream>>>(ws + O_UBUF, 0.0f, (long)U_*EMB);
    fill_kernel<<<cdivl((long)B_*EMB,   T), T, 0, stream>>>(ws + O_BBUF, 0.0f, (long)B_*EMB);
    fill_kernel<<<1, T, 0, stream>>>(out + OUT_REG, 0.0f, 2);  // reg_loss, diff_loss

    // ---- pack weights (tiny, once per call) ----
    pack_w_kernel<<<cdivl((long)64*64,   T), T, 0, stream>>>(W_gat, wsb + PB_WGP, 64, 64);
    pack_w_kernel<<<cdivl((long)192*256, T), T, 0, stream>>>(W1,    wsb + PB_W1P, 192, 256);
    pack_w_kernel<<<cdivl((long)256*64,  T), T, 0, stream>>>(W2,    wsb + PB_W2P, 256, 64);

    // ---- GAT: Wx = embed0 @ W_gat ----
    concat_pack_kernel<<<cdivl((long)N_*EMB, T), T, 0, stream>>>(
        users_feature, bundles_feature, wsb + PB_E0P, (long)U_*EMB, (long)N_*EMB);
    {
        dim3 g(cdivl(N_, 64), EMB / 64);
        gemm_wmma_packed<<<g, 128, 0, stream>>>(wsb + PB_E0P, wsb + PB_WGP, nullptr,
                                                ws + O_WX, nullptr, N_, EMB, EMB, 0);
    }
    elr_kernel<<<cdivl((long)N_*32, T), T, 0, stream>>>(
        ws + O_WX, attn_l, attn_r, ws + O_EL, ws + O_ER, N_);

    // ---- GAT edge softmax (3 passes) ----
    gat_edge_e<<<cdivl(Eg, T), T, 0, stream>>>(
        ws + O_EL, ws + O_ER, g_src, g_dst, ws + O_EX, ws + O_M, Eg);
    gat_edge_ex<<<cdivl(Eg, T), T, 0, stream>>>(
        ws + O_EX, g_dst, ws + O_M, ws + O_S, Eg);
    gat_edge_scatter<<<cdivl(Eg * EMB, T), T, 0, stream>>>(
        ws + O_EX, ws + O_S, g_src, g_dst, ws + O_WX, ws + O_GAT, Eg);

    // ---- embed_r = spmm(r, gat_out + bias_gat) ----
    spmm_scatter<<<cdivl(Er * EMB, T), T, 0, stream>>>(
        r_row, r_col, r_val, ws + O_GAT, bias_gat, ws + O_EMBR, Er);

    // ---- item propagation ----
    spmm_scatter<<<cdivl(Eai * EMB, T), T, 0, stream>>>(
        Ai_row, Ai_col, Ai_val, items_feature, nullptr, ws + O_IUI, Eai);
    spmm_scatter<<<cdivl(Ebi * EMB, T), T, 0, stream>>>(
        Bi_row, Bi_col, Bi_val, items_feature, nullptr, ws + O_IBI, Ebi);
    items_combine<<<cdivl((long)I_*EMB, T), T, 0, stream>>>(
        ws + O_IUI, ws + O_IBI, items_feature, ws + O_ITEMSF, (long)I_*EMB);

    // ---- user / bundle aggregation ----
    spmm_scatter<<<cdivl(Eui * EMB, T), T, 0, stream>>>(
        ui_row, ui_col, ui_val, ws + O_ITEMSF, nullptr, ws + O_UBUF, Eui);
    spmm_scatter<<<cdivl(Ebb * EMB, T), T, 0, stream>>>(
        bi_row, bi_col, bi_val, ws + O_ITEMSF, nullptr, ws + O_BBUF, Ebb);
    relu_add<<<cdivl((long)U_*EMB, T), T, 0, stream>>>(
        ws + O_UBUF, ws + O_EMBR, ws + O_USERSF, (long)U_*EMB);                   // users_on_ub
    relu_add<<<cdivl((long)B_*EMB, T), T, 0, stream>>>(
        ws + O_BBUF, ws + O_EMBR + (long)U_*EMB, ws + O_BUNDF, (long)B_*EMB);     // bundles_on_ub

    // ---- diffusion MLP ----
    diffusion_prep_pack<<<cdivl((long)U_*3*EMB, T), T, 0, stream>>>(
        ws + O_EMBR /* x_start = users_on_ub */, noise, ws + O_USERSF, t,
        wsb + PB_HINP, (long)U_*3*EMB);
    {
        dim3 g(cdivl(U_, 64), HID / 64);
        gemm_wmma_packed<<<g, 128, 0, stream>>>(wsb + PB_HINP, wsb + PB_W1P, b1,
                                                nullptr, wsb + PB_H1P,
                                                U_, 3 * EMB, HID, /*tanh*/ 1);
    }
    {
        dim3 g(cdivl(U_, 64), EMB / 64);
        gemm_wmma_packed<<<g, 128, 0, stream>>>(wsb + PB_H1P, wsb + PB_W2P, b2,
                                                ws + O_PREDX, nullptr,
                                                U_, HID, EMB, 0);
    }

    // ---- losses ----
    diffloss_kernel<<<cdivl((long)U_*EMB, T), T, 0, stream>>>(
        ws + O_PREDX, ws + O_EMBR, (long)U_*EMB, 1.0f / ((float)U_ * EMB), out + OUT_DIFF);
    sqsum_kernel<<<cdivl((long)U_*EMB, T), T, 0, stream>>>(
        ws + O_USERSF, (long)U_*EMB, L2NORM, out + OUT_REG);
    sqsum_kernel<<<cdivl((long)B_*EMB, T), T, 0, stream>>>(
        ws + O_BUNDF, (long)B_*EMB, L2NORM, out + OUT_REG);
    sqsum_kernel<<<cdivl((long)I_*EMB, T), T, 0, stream>>>(
        ws + O_ITEMSF, (long)I_*EMB, L2NORM, out + OUT_REG);

    // ---- scores ----
    pred_kernel<<<cdivl((long)BATCH * NB * 32, T), T, 0, stream>>>(
        ws + O_USERSF, ws + O_BUNDF, users, bundles, out + OUT_PRED);
    bound_kernel<<<cdivl((long)BATCH * 32, T), T, 0, stream>>>(
        ws + O_PREDX, user_bound, users, out + OUT_BOUND);

    // ---- users_feature passthrough ----
    copy_kernel<<<cdivl((long)U_*EMB, T), T, 0, stream>>>(
        users_feature, out + OUT_UF, (long)U_*EMB);
}
